// CausalSelfAttention_36953898614824
// MI455X (gfx1250) — matrix operand order repaired and checked by reading, not compiled, not run on hardware
//
#include <hip/hip_runtime.h>
#include <stddef.h>

#define BATCH 2
#define TSEQ  2048
#define CDIM  1024
#define HEADS 16
#define DHEAD 64
#define C3    (3*CDIM)
#define ROWS  (BATCH*TSEQ)   // 4096

typedef __attribute__((ext_vector_type(16))) __bf16        v16bf;
typedef __attribute__((ext_vector_type(8)))  float         v8f;
typedef __attribute__((ext_vector_type(4)))  unsigned int  u32x4;
typedef __attribute__((ext_vector_type(2)))  unsigned int  u32x2;
typedef __attribute__((ext_vector_type(4)))  float         f32x4;

union Frag {
  v16bf bf;
  u32x4 q4[2];
  unsigned int u[8];
};

__device__ __forceinline__ unsigned short f2bf(float f) {
  unsigned int u = __float_as_uint(f);
  u += 0x7fffu + ((u >> 16) & 1u);          // round-to-nearest-even
  return (unsigned short)(u >> 16);
}
__device__ __forceinline__ unsigned int pack2bf(float lo, float hi) {
  return (unsigned int)f2bf(lo) | ((unsigned int)f2bf(hi) << 16);
}
__device__ __forceinline__ float lane_gather(float v, int srcLane) {
  return __int_as_float(__builtin_amdgcn_ds_bpermute(srcLane << 2, __float_as_int(v)));
}
__device__ __forceinline__ v8f wmma_bf16(v16bf a, v16bf b, v8f c) {
  return __builtin_amdgcn_wmma_f32_16x16x32_bf16(false, a, false, b, (short)0, c,
                                                 false, false);
}
__device__ __forceinline__ void load_afrag(Frag& f, const unsigned short* p) {
  const u32x4* q = reinterpret_cast<const u32x4*>(p);
  f.q4[0] = q[0]; f.q4[1] = q[2];          // K chunks +0 and +16 elements
}
__device__ __forceinline__ void load_bfrag(Frag& f, const unsigned short* p) {
  const u32x4* q = reinterpret_cast<const u32x4*>(p);
  f.q4[0] = q[0]; f.q4[1] = q[2];          // K chunks +0 and +16 elements
}

// ---------------- cast kernels ----------------
__global__ __launch_bounds__(256) void cast_x_kernel(const float* __restrict__ x,
                                                     unsigned short* __restrict__ xb,
                                                     int n4) {
  int i = blockIdx.x * blockDim.x + threadIdx.x;
  if (i >= n4) return;
  f32x4 f = reinterpret_cast<const f32x4*>(x)[i];
  u32x2 o;
  o.x = pack2bf(f.x, f.y);
  o.y = pack2bf(f.z, f.w);
  reinterpret_cast<u32x2*>(xb)[i] = o;
}

// Wt[n*K + k] = W[k*N + n], bf16
__global__ __launch_bounds__(256) void cast_transpose_kernel(const float* __restrict__ W,
                                                             unsigned short* __restrict__ Wt,
                                                             int K, int N) {
  int idx = blockIdx.x * blockDim.x + threadIdx.x;
  if (idx >= K * N) return;
  int n = idx / K;
  int k = idx - n * K;
  Wt[idx] = f2bf(W[(size_t)k * N + n]);
}

// ---------------- shared 32x64 GEMM core (2 row tiles x 4 col tiles) ----------------
__device__ __forceinline__ void gemm_core_32x64(const unsigned short* __restrict__ A,
                                                const unsigned short* __restrict__ Bt,
                                                int rowBase, int colBase,
                                                int li, int hi, v8f acc[2][4]) {
  const unsigned short* a0p = A + (size_t)(rowBase + li) * CDIM + hi * 8;
  const unsigned short* a1p = A + (size_t)(rowBase + 16 + li) * CDIM + hi * 8;
  const unsigned short* b0p = Bt + (size_t)(colBase + li) * CDIM + hi * 8;
  const unsigned short* b1p = b0p + (size_t)16 * CDIM;
  const unsigned short* b2p = b0p + (size_t)32 * CDIM;
  const unsigned short* b3p = b0p + (size_t)48 * CDIM;

  for (int kk = 0; kk < CDIM; kk += 32) {
    Frag a0, a1, b0, b1, b2, b3;
    load_afrag(a0, a0p + kk);
    load_afrag(a1, a1p + kk);
    load_bfrag(b0, b0p + kk);
    load_bfrag(b1, b1p + kk);
    load_bfrag(b2, b2p + kk);
    load_bfrag(b3, b3p + kk);
    // prefetch a few k-steps ahead (global_prefetch_b8; harmless past the end)
    __builtin_prefetch(a0p + kk + 128, 0, 0);
    __builtin_prefetch(a1p + kk + 128, 0, 0);
    __builtin_prefetch(b0p + kk + 128, 0, 0);
    __builtin_prefetch(b2p + kk + 128, 0, 0);

    acc[0][0] = wmma_bf16(a0.bf, b0.bf, acc[0][0]);
    acc[1][0] = wmma_bf16(a1.bf, b0.bf, acc[1][0]);
    acc[0][1] = wmma_bf16(a0.bf, b1.bf, acc[0][1]);
    acc[1][1] = wmma_bf16(a1.bf, b1.bf, acc[1][1]);
    acc[0][2] = wmma_bf16(a0.bf, b2.bf, acc[0][2]);
    acc[1][2] = wmma_bf16(a1.bf, b2.bf, acc[1][2]);
    acc[0][3] = wmma_bf16(a0.bf, b3.bf, acc[0][3]);
    acc[1][3] = wmma_bf16(a1.bf, b3.bf, acc[1][3]);
  }
}

// ---------------- QKV GEMM: qkv = x @ W_attn + b, scatter to q/k/vT ----------------
__global__ __launch_bounds__(128) void gemm_qkv_kernel(
    const unsigned short* __restrict__ xb,   // 4096 x 1024 bf16 row-major
    const unsigned short* __restrict__ wat,  // 3072 x 1024 bf16 (transposed W_attn)
    const float* __restrict__ bias,          // 3072
    unsigned short* __restrict__ qo,         // (B,H,T,D)
    unsigned short* __restrict__ ko,         // (B,H,T,D)
    unsigned short* __restrict__ vto) {      // (B,H,D,T)
  const int lane = threadIdx.x & 31, wave = threadIdx.x >> 5;
  const int li = lane & 15, hi = lane >> 4;
  const int rowBase = (blockIdx.x * 4 + wave) * 32;
  const int colBase = blockIdx.y * 64;
  v8f acc[2][4] = {};

  gemm_core_32x64(xb, wat, rowBase, colBase, li, hi, acc);

#pragma unroll
  for (int rt = 0; rt < 2; ++rt) {
#pragma unroll
    for (int nt = 0; nt < 4; ++nt) {
#pragma unroll
      for (int r = 0; r < 8; ++r) {
        int row = rowBase + rt * 16 + r + 8 * hi;
        int col = colBase + nt * 16 + li;
        unsigned short bv = f2bf(acc[rt][nt][r] + bias[col]);
        int which = col >> 10;          // uniform per wave (strip within q/k/v)
        int c = col & (CDIM - 1);
        int h = c >> 6, d = c & 63;
        int b = row >> 11, t = row & (TSEQ - 1);
        size_t bh = (size_t)b * HEADS + h;
        if (which == 0)      qo[(bh * TSEQ + t) * DHEAD + d] = bv;
        else if (which == 1) ko[(bh * TSEQ + t) * DHEAD + d] = bv;
        else                 vto[(bh * DHEAD + d) * TSEQ + t] = bv;
      }
    }
  }
}

// ---------------- flash attention, one wave per 16-query tile ----------------
__global__ __launch_bounds__(128) void attn_kernel(
    const unsigned short* __restrict__ qb,
    const unsigned short* __restrict__ kbuf,
    const unsigned short* __restrict__ vtb,
    unsigned short* __restrict__ yb) {   // (B,T,H,D) bf16
  const int lane = threadIdx.x & 31, wave = threadIdx.x >> 5;
  const int li = lane & 15, hi = lane >> 4;
  const int qtile = blockIdx.x * 4 + wave;      // 0..127
  const int bh = blockIdx.y;                    // 0..31
  const int qbase = qtile * 16;
  const int qi = qbase + li;

  const unsigned short* Q  = qb   + (size_t)bh * TSEQ * DHEAD;
  const unsigned short* Kp = kbuf + (size_t)bh * TSEQ * DHEAD;
  const unsigned short* VT = vtb  + (size_t)bh * DHEAD * TSEQ;

  // Q^T B-fragments, reused for every KV step (d-chunks 0..31 and 32..63)
  Frag qf[2];
#pragma unroll
  for (int ds = 0; ds < 2; ++ds)
    load_bfrag(qf[ds], Q + (size_t)qi * DHEAD + ds * 32 + hi * 8);

  v8f acc[4] = {};                // O^T accumulators: 4 d-tiles x (8 VGPR) f32
  float m_i = -1e30f, l_i = 0.f;
  const float scale = 0.125f;     // 1/sqrt(64)

  for (int kb = 0; kb <= qbase + 15; kb += 32) {
    // load all 4 K A-fragments first, then issue the 4 S^T WMMAs back-to-back
    Frag kf00, kf10, kf01, kf11;  // kf[tile][ds]
    load_afrag(kf00, Kp + (size_t)(kb + li) * DHEAD + hi * 8);
    load_afrag(kf10, Kp + (size_t)(kb + 16 + li) * DHEAD + hi * 8);
    load_afrag(kf01, Kp + (size_t)(kb + li) * DHEAD + 32 + hi * 8);
    load_afrag(kf11, Kp + (size_t)(kb + 16 + li) * DHEAD + 32 + hi * 8);

    v8f s0 = {0,0,0,0,0,0,0,0};
    v8f s1 = {0,0,0,0,0,0,0,0};
    s0 = wmma_bf16(kf00.bf, qf[0].bf, s0);
    s1 = wmma_bf16(kf10.bf, qf[0].bf, s1);
    s0 = wmma_bf16(kf01.bf, qf[1].bf, s0);
    s1 = wmma_bf16(kf11.bf, qf[1].bf, s1);

    // issue V^T fragment loads early; softmax VALU below hides their latency
    Frag vf0, vf1, vf2, vf3;
    load_afrag(vf0, VT + (size_t)(li) * TSEQ + kb + hi * 8);
    load_afrag(vf1, VT + (size_t)(16 + li) * TSEQ + kb + hi * 8);
    load_afrag(vf2, VT + (size_t)(32 + li) * TSEQ + kb + hi * 8);
    load_afrag(vf3, VT + (size_t)(48 + li) * TSEQ + kb + hi * 8);

    // scale + causal mask + streaming softmax (keys live in-lane + partner lane)
    float p0v[8], p1v[8];
    float mloc = -1e30f;
#pragma unroll
    for (int r = 0; r < 8; ++r) {
      int key0 = kb + r + 8 * hi;
      float v0 = s0[r] * scale;  v0 = (key0 <= qi) ? v0 : -1e30f;
      float v1 = s1[r] * scale;  v1 = (key0 + 16 <= qi) ? v1 : -1e30f;
      p0v[r] = v0; p1v[r] = v1;
      mloc = fmaxf(mloc, fmaxf(v0, v1));
    }
    mloc = fmaxf(mloc, lane_gather(mloc, lane ^ 16));
    float mnew  = fmaxf(m_i, mloc);
    float alpha = __expf(m_i - mnew);
    float lloc  = 0.f;
#pragma unroll
    for (int r = 0; r < 8; ++r) {
      p0v[r] = __expf(p0v[r] - mnew); lloc += p0v[r];
      p1v[r] = __expf(p1v[r] - mnew); lloc += p1v[r];
    }
    lloc += lane_gather(lloc, lane ^ 16);
    l_i = l_i * alpha + lloc;
    m_i = mnew;
#pragma unroll
    for (int dt = 0; dt < 4; ++dt)
#pragma unroll
      for (int r = 0; r < 8; ++r) acc[dt][r] *= alpha;

    // Rebuild P^T as a 32x16 B-fragment from the C-layout tiles via bpermute.
    // B-frag element e at lane l: key = 8*(l>>4) + (e&7) + 16*(e>>3); source: tile (e>>3),
    // row 8*(l>>4) + (e&7) held at lane (l&15) + 16*(l>>4).
    float elo[8], ehi[8];
#pragma unroll
    for (int r = 0; r < 8; ++r) {
      float t0a = lane_gather(p0v[r], li);
      float t0b = lane_gather(p0v[r], li + 16);
      float t1a = lane_gather(p1v[r], li);
      float t1b = lane_gather(p1v[r], li + 16);
      elo[r] = hi ? t0b : t0a;
      ehi[r] = hi ? t1b : t1a;
    }
    Frag pf;
#pragma unroll
    for (int j = 0; j < 4; ++j) {
      pf.u[j]     = pack2bf(elo[2 * j], elo[2 * j + 1]);
      pf.u[4 + j] = pack2bf(ehi[2 * j], ehi[2 * j + 1]);
    }

    // O^T += V^T . P^T   (K-dim = 32 keys, one WMMA per 16-row d-tile)
    acc[0] = wmma_bf16(vf0.bf, pf.bf, acc[0]);
    acc[1] = wmma_bf16(vf1.bf, pf.bf, acc[1]);
    acc[2] = wmma_bf16(vf2.bf, pf.bf, acc[2]);
    acc[3] = wmma_bf16(vf3.bf, pf.bf, acc[3]);
  }

  float inv = 1.0f / l_i;
  int b = bh >> 4, h = bh & (HEADS - 1);
#pragma unroll
  for (int dt = 0; dt < 4; ++dt)
#pragma unroll
    for (int r = 0; r < 8; ++r) {
      int d = dt * 16 + r + 8 * hi;
      size_t off = (((size_t)b * TSEQ + qi) * HEADS + h) * DHEAD + d;
      yb[off] = f2bf(acc[dt][r] * inv);
    }
}

// ---------------- projection GEMM: out = y @ W_proj + b (fp32 out) ----------------
__global__ __launch_bounds__(128) void gemm_proj_kernel(
    const unsigned short* __restrict__ ybuf,  // 4096 x 1024 bf16
    const unsigned short* __restrict__ wpt,   // 1024 x 1024 bf16 (transposed)
    const float* __restrict__ bias,
    float* __restrict__ out) {
  const int lane = threadIdx.x & 31, wave = threadIdx.x >> 5;
  const int li = lane & 15, hi = lane >> 4;
  const int rowBase = (blockIdx.x * 4 + wave) * 32;
  const int colBase = blockIdx.y * 64;
  v8f acc[2][4] = {};

  gemm_core_32x64(ybuf, wpt, rowBase, colBase, li, hi, acc);

#pragma unroll
  for (int rt = 0; rt < 2; ++rt) {
#pragma unroll
    for (int nt = 0; nt < 4; ++nt) {
#pragma unroll
      for (int r = 0; r < 8; ++r) {
        int row = rowBase + rt * 16 + r + 8 * hi;
        int col = colBase + nt * 16 + li;
        out[(size_t)row * CDIM + col] = acc[rt][nt][r] + bias[col];
      }
    }
  }
}

// ---------------- launch ----------------
extern "C" void kernel_launch(void* const* d_in, const int* in_sizes, int n_in,
                              void* d_out, int out_size, void* d_ws, size_t ws_size,
                              hipStream_t stream) {
  (void)in_sizes; (void)n_in; (void)out_size; (void)ws_size;
  const float* x  = (const float*)d_in[0];
  const float* Wa = (const float*)d_in[1];
  const float* ba = (const float*)d_in[2];
  const float* Wp = (const float*)d_in[3];
  const float* bp = (const float*)d_in[4];
  float* out = (float*)d_out;

  char* ws = (char*)d_ws;
  unsigned short* xb  = (unsigned short*)(ws);                        //  8 MiB
  unsigned short* wat = (unsigned short*)(ws + (size_t)( 8u << 20));  //  6 MiB
  unsigned short* wpt = (unsigned short*)(ws + (size_t)(14u << 20));  //  2 MiB
  unsigned short* qb  = (unsigned short*)(ws + (size_t)(16u << 20));  //  8 MiB
  unsigned short* kb  = (unsigned short*)(ws + (size_t)(24u << 20));  //  8 MiB
  unsigned short* vtb = (unsigned short*)(ws + (size_t)(32u << 20));  //  8 MiB
  unsigned short* yb  = (unsigned short*)(ws + (size_t)(40u << 20));  //  8 MiB (48 MiB total)

  int n4 = ROWS * CDIM / 4;
  cast_x_kernel<<<n4 / 256, 256, 0, stream>>>(x, xb, n4);
  cast_transpose_kernel<<<(CDIM * C3 + 255) / 256, 256, 0, stream>>>(Wa, wat, CDIM, C3);
  cast_transpose_kernel<<<(CDIM * CDIM + 255) / 256, 256, 0, stream>>>(Wp, wpt, CDIM, CDIM);

  gemm_qkv_kernel<<<dim3(ROWS / 128, C3 / 64), 128, 0, stream>>>(xb, wat, ba, qb, kb, vtb);
  attn_kernel<<<dim3(TSEQ / 64, BATCH * HEADS), 128, 0, stream>>>(qb, kb, vtb, yb);
  gemm_proj_kernel<<<dim3(ROWS / 128, CDIM / 64), 128, 0, stream>>>(yb, wpt, bp, out);
}
